// Multi_Head_Attention_36447092474267
// MI455X (gfx1250) — matrix operand order repaired and checked by reading, not compiled, not run on hardware
//
#include <hip/hip_runtime.h>

// ---------- types ----------
typedef __attribute__((ext_vector_type(16))) _Float16 v16h;
typedef __attribute__((ext_vector_type(8)))  _Float16 v8h;
typedef __attribute__((ext_vector_type(8)))  float    v8f;

#define B_DIM 2
#define S_DIM 2048
#define HID   1024
#define NH    16
#define DH    64

// GEMM staging: k-chunk = 64 halves, LDS row padded 64 -> 72 halves
// (144B row stride => bank stride 36, gcd(36,64)=4 => 16 frag lanes on 16 banks)
#define KC        64
#define BT_STRIDE 72
#define BT_BUF    (64 * BT_STRIDE)

// ---------- WMMA helper (CDNA5 16x16x32 f16 -> f32) ----------
__device__ __forceinline__ v8f wmma_f16(v16h a, v16h b, v8f c) {
  return __builtin_amdgcn_wmma_f32_16x16x32_f16(false, a, false, b, (short)0, c, false, false);
}

// A fragment (16x32 f16): lane holds row M=lane%16; K=(lane/16)*16+j contiguous.
// 'p' points at  base + (lane%16)*ld + (lane>>4)*8; helper adds another (lane>>4)*8.
__device__ __forceinline__ v16h load_frag_a(const _Float16* p) {
  p += ((threadIdx.x & 31) >> 4) * 8;
  union { v16h v; v8h h[2]; } u;
  u.h[0] = *reinterpret_cast<const v8h*>(p);
  u.h[1] = *reinterpret_cast<const v8h*>(p + 8);
  return u.v;
}

// B fragment (32x16 f16): lane holds col N=lane%16; K=(lane/16)*16+j contiguous.
// 'p' points at base_of_column + (lane>>4)*16.
__device__ __forceinline__ v16h load_frag_b(const _Float16* p) {
  union { v16h v; v8h h[2]; } u;
  u.h[0] = *reinterpret_cast<const v8h*>(p);
  u.h[1] = *reinterpret_cast<const v8h*>(p + 8);
  return u.v;
}

// Async per-lane 16B copy global -> LDS (tracked on ASYNCcnt).
__device__ __forceinline__ void async_ld_b128(unsigned lds_off, const _Float16* gptr) {
  asm volatile("global_load_async_to_lds_b128 %0, %1, off"
               :: "v"(lds_off), "v"(gptr) : "memory");
}

// Raw split workgroup barrier WITHOUT the loadcnt-draining fence of
// __syncthreads(); ordering is handled by explicit async/ds waits.
__device__ __forceinline__ void wg_barrier() {
  asm volatile("s_barrier_signal -1\n\ts_barrier_wait -1" ::: "memory");
}

// ---------- kernel 1: fp32 -> f16 copy of x ----------
__global__ __launch_bounds__(256) void k_cvt_x(const float* __restrict__ x,
                                               _Float16* __restrict__ xh, int n) {
  int i = (blockIdx.x * 256 + threadIdx.x) * 4;
  if (i + 3 < n) {
    float4 v = *reinterpret_cast<const float4*>(x + i);
    xh[i + 0] = (_Float16)v.x;
    xh[i + 1] = (_Float16)v.y;
    xh[i + 2] = (_Float16)v.z;
    xh[i + 3] = (_Float16)v.w;
  }
}

// ---------- kernel 2: fp32 W[K][N] -> f16 Wt[N][K] (tiled transpose) ----------
__global__ __launch_bounds__(256) void k_cvt_wT(const float* __restrict__ W,
                                                _Float16* __restrict__ Wt) {
  __shared__ float tile[32][33];
  int kb = blockIdx.y * 32, nb = blockIdx.x * 32;
  int tx = threadIdx.x, ty = threadIdx.y;           // 32 x 8
#pragma unroll
  for (int i = 0; i < 4; ++i)
    tile[ty + 8 * i][tx] = W[(kb + ty + 8 * i) * HID + nb + tx];
  __syncthreads();
#pragma unroll
  for (int i = 0; i < 4; ++i)
    Wt[(nb + ty + 8 * i) * HID + kb + tx] = (_Float16)tile[tx][ty + 8 * i];
}

// ---------- kernel 3: FUSED QKV projection GEMM ----------
// Block: 256 threads = 8 waves, tile 128 rows x 64 cols, all three weight
// matrices staged per stage (6 async B128/thread, 55KB LDS, 24 WMMA/stage).
// One pair of A fragments feeds Q, K and V accumulators.
__global__ __launch_bounds__(256) void k_gemm_qkv(
    const _Float16* __restrict__ xh,
    const _Float16* __restrict__ wqT, const _Float16* __restrict__ wkT,
    const _Float16* __restrict__ wvT,
    const float* __restrict__ bq, const float* __restrict__ bk,
    const float* __restrict__ bv,
    _Float16* __restrict__ Qb, _Float16* __restrict__ Kb,
    _Float16* __restrict__ Vt) {
  __shared__ __align__(16) _Float16 bt[6 * BT_BUF];   // [mode][buf] tiles
  const int lane = threadIdx.x & 31, wave = threadIdx.x >> 5;
  const int row = lane & 15, hf = lane >> 4;
  const int tid = threadIdx.x;
  const int m0 = blockIdx.y * 128 + wave * 16;
  const int n0 = blockIdx.x * 64;

  const int nl = tid >> 2;              // 0..63 : local n
  const int kc = (tid & 3) * 8;         // 0,8,16,24 halves within half-chunk
  const _Float16* gw[3] = { wqT + (n0 + nl) * HID + kc,
                            wkT + (n0 + nl) * HID + kc,
                            wvT + (n0 + nl) * HID + kc };
  unsigned ldo[3][2][2];
#pragma unroll
  for (int w = 0; w < 3; ++w)
#pragma unroll
    for (int bu = 0; bu < 2; ++bu) {
      _Float16* p = bt + (w * 2 + bu) * BT_BUF + nl * BT_STRIDE + kc;
      ldo[w][bu][0] = (unsigned)(uintptr_t)p;
      ldo[w][bu][1] = (unsigned)(uintptr_t)(p + 32);
    }
  const _Float16* aptr = xh + (m0 + row) * HID + hf * 8;

#pragma unroll
  for (int w = 0; w < 3; ++w) {
    async_ld_b128(ldo[w][0][0], gw[w]);
    async_ld_b128(ldo[w][0][1], gw[w] + 32);
  }
  v16h a0 = load_frag_a(aptr);
  v16h a1 = load_frag_a(aptr + 32);

  v8f acc[3][4] = {};
  int cur = 0;
  for (int k0 = 0; k0 < HID; k0 += KC) {
    const bool hasNext = (k0 + KC) < HID;
    if (hasNext) {
#pragma unroll
      for (int w = 0; w < 3; ++w) {
        async_ld_b128(ldo[w][cur ^ 1][0], gw[w] + k0 + KC);
        async_ld_b128(ldo[w][cur ^ 1][1], gw[w] + k0 + KC + 32);
      }
    }
    v16h an0 = a0, an1 = a1;
    if (hasNext) {
      an0 = load_frag_a(aptr + k0 + KC);
      an1 = load_frag_a(aptr + k0 + KC + 32);
    }
    if (hasNext) asm volatile("s_wait_asynccnt 0x6" ::: "memory");
    else         asm volatile("s_wait_asynccnt 0x0" ::: "memory");
    wg_barrier();                        // buffers [*][cur] resident, all waves
#pragma unroll
    for (int w = 0; w < 3; ++w) {
      const _Float16* bbuf = bt + (w * 2 + cur) * BT_BUF;
      v16h bf[8];
#pragma unroll
      for (int t = 0; t < 4; ++t)
        bf[t] = load_frag_b(bbuf + (t * 16 + row) * BT_STRIDE + hf * 16);
#pragma unroll
      for (int t = 0; t < 4; ++t)
        bf[4 + t] = load_frag_b(bbuf + (t * 16 + row) * BT_STRIDE + 32 + hf * 16);
#pragma unroll
      for (int t = 0; t < 4; ++t) acc[w][t] = wmma_f16(a0, bf[t], acc[w][t]);
#pragma unroll
      for (int t = 0; t < 4; ++t) acc[w][t] = wmma_f16(a1, bf[4 + t], acc[w][t]);
    }
    wg_barrier();                        // reads done before buffer rewrite
    a0 = an0; a1 = an1;
    cur ^= 1;
  }

  const float QSCALE = 0.125f * 1.44269504088896340736f;  // (1/sqrt(64))*log2(e)
#pragma unroll
  for (int t = 0; t < 4; ++t) {
    const int n = n0 + t * 16 + row;
    const int h = n >> 6, d = n & 63;
    const float bnq = bq[n], bnk = bk[n], bnv = bv[n];
#pragma unroll
    for (int r = 0; r < 8; ++r) {
      const int m = m0 + r + 8 * hf;
      const int b = m >> 11, s = m & 2047;
      Qb[(((b * NH + h) * S_DIM + s) << 6) + d] =
          (_Float16)((acc[0][t][r] + bnq) * QSCALE);
      Kb[(((b * NH + h) * S_DIM + s) << 6) + d] = (_Float16)(acc[1][t][r] + bnk);
      Vt[(((b * NH + h) * DH + d) << 11) + s]   = (_Float16)(acc[2][t][r] + bnv);
    }
  }
}

// ---------- shared GEMM core (single weight; used by output projection) -----
__device__ __forceinline__ void gemm_core(const _Float16* __restrict__ ah,
                                          const _Float16* __restrict__ wT,
                                          _Float16* bt,   // shared, 2*BT_BUF halves
                                          int m0, int n0, v8f acc[4]) {
  const int lane = threadIdx.x & 31;
  const int row = lane & 15, hf = lane >> 4;
  const int tid = threadIdx.x;
  const int nl = tid >> 2;
  const int kc = (tid & 3) * 8;
  const _Float16* gsrc = wT + (n0 + nl) * HID + kc;
  _Float16* l0 = bt + nl * BT_STRIDE + kc;
  const unsigned ld0a = (unsigned)(uintptr_t)(l0);
  const unsigned ld0b = (unsigned)(uintptr_t)(l0 + 32);
  const unsigned ld1a = (unsigned)(uintptr_t)(l0 + BT_BUF);
  const unsigned ld1b = (unsigned)(uintptr_t)(l0 + BT_BUF + 32);
  const _Float16* aptr = ah + (m0 + row) * HID + hf * 8;

  async_ld_b128(ld0a, gsrc);
  async_ld_b128(ld0b, gsrc + 32);
  v16h a0 = load_frag_a(aptr);
  v16h a1 = load_frag_a(aptr + 32);
  int cur = 0;
  for (int k0 = 0; k0 < HID; k0 += KC) {
    const bool hasNext = (k0 + KC) < HID;
    if (hasNext) {
      const _Float16* gnext = gsrc + k0 + KC;
      async_ld_b128(cur ? ld0a : ld1a, gnext);
      async_ld_b128(cur ? ld0b : ld1b, gnext + 32);
    }
    v16h an0 = a0, an1 = a1;
    if (hasNext) {
      an0 = load_frag_a(aptr + k0 + KC);
      an1 = load_frag_a(aptr + k0 + KC + 32);
    }
    if (hasNext) asm volatile("s_wait_asynccnt 0x2" ::: "memory");
    else         asm volatile("s_wait_asynccnt 0x0" ::: "memory");
    wg_barrier();
    const _Float16* bbuf = bt + cur * BT_BUF;
    v16h bf[8];
#pragma unroll
    for (int t = 0; t < 4; ++t)
      bf[t] = load_frag_b(bbuf + (t * 16 + row) * BT_STRIDE + hf * 16);
#pragma unroll
    for (int t = 0; t < 4; ++t)
      bf[4 + t] = load_frag_b(bbuf + (t * 16 + row) * BT_STRIDE + 32 + hf * 16);
#pragma unroll
    for (int t = 0; t < 4; ++t) acc[t] = wmma_f16(a0, bf[t], acc[t]);
#pragma unroll
    for (int t = 0; t < 4; ++t) acc[t] = wmma_f16(a1, bf[4 + t], acc[t]);
    wg_barrier();
    a0 = an0; a1 = an1;
    cur ^= 1;
  }
}

// ---------- kernel 4: flash attention (exp2 domain, K/V double buffered) ----
// grid = (S/64, B*NH), block = 128 (4 waves); each wave: 16 query rows.
__global__ __launch_bounds__(128) void k_attn(const _Float16* __restrict__ Qb,
                                              const _Float16* __restrict__ Kb,
                                              const _Float16* __restrict__ Vt,
                                              _Float16* __restrict__ ctx) {
  __shared__ __align__(16) _Float16 plds[4][16 * 32];   // P tile per wave
  const int lane = threadIdx.x & 31, wave = threadIdx.x >> 5;
  const int row = lane & 15, hf = lane >> 4;
  const int bh = blockIdx.y, b = bh >> 4, h = bh & 15;
  const int q0 = blockIdx.x * 64 + wave * 16;

  const _Float16* Qh = Qb + (size_t)bh * S_DIM * DH;
  const _Float16* Kh = Kb + (size_t)bh * S_DIM * DH;
  const _Float16* Vh = Vt + (size_t)bh * DH * S_DIM;

  const v16h aQ0 = load_frag_a(Qh + (q0 + row) * DH + hf * 8);
  const v16h aQ1 = load_frag_a(Qh + (q0 + row) * DH + 32 + hf * 8);

  v8f o[4] = {};
  float m8[8], l8[8];
#pragma unroll
  for (int r = 0; r < 8; ++r) { m8[r] = -3.0e38f; l8[r] = 0.0f; }

  _Float16* pl = plds[wave];

  // ---- preload kv-chunk 0 fragments ----
  v16h kcur[4], vcur[4];
  {
    const _Float16* kp0 = Kh + row * DH + hf * 16;
    const _Float16* kp1 = Kh + (16 + row) * DH + hf * 16;
    kcur[0] = load_frag_b(kp0);     kcur[1] = load_frag_b(kp0 + 32);
    kcur[2] = load_frag_b(kp1);     kcur[3] = load_frag_b(kp1 + 32);
#pragma unroll
    for (int t = 0; t < 4; ++t)
      vcur[t] = load_frag_b(Vh + (t * 16 + row) * S_DIM + hf * 16);
  }

  for (int kv0 = 0; kv0 < S_DIM; kv0 += 32) {
    const bool hasNext = (kv0 + 32) < S_DIM;
    // ---- prefetch kv-chunk i+1 (in flight through softmax + PV) ----
    v16h kn[4], vn[4];
    if (hasNext) {
      const int kvn = kv0 + 32;
      const _Float16* kp0 = Kh + (kvn + row) * DH + hf * 16;
      const _Float16* kp1 = Kh + (kvn + 16 + row) * DH + hf * 16;
      kn[0] = load_frag_b(kp0);     kn[1] = load_frag_b(kp0 + 32);
      kn[2] = load_frag_b(kp1);     kn[3] = load_frag_b(kp1 + 32);
#pragma unroll
      for (int t = 0; t < 4; ++t)
        vn[t] = load_frag_b(Vh + (t * 16 + row) * S_DIM + kvn + hf * 16);
    } else {
#pragma unroll
      for (int t = 0; t < 4; ++t) { kn[t] = kcur[t]; vn[t] = vcur[t]; }
    }

    // ---- scores: S = Q (16x64) * K^T (64x32) (uses *old* loads) ----
    v8f s[2];
    {
      v8f z0 = {}, z1 = {};
      z0 = wmma_f16(aQ0, kcur[0], z0);
      z0 = wmma_f16(aQ1, kcur[1], z0);
      z1 = wmma_f16(aQ0, kcur[2], z1);
      z1 = wmma_f16(aQ1, kcur[3], z1);
      s[0] = z0; s[1] = z1;
    }
    // ---- streaming softmax in exp2 domain ----
#pragma unroll
    for (int r = 0; r < 8; ++r) {
      float mx = fmaxf(s[0][r], s[1][r]);
      mx = fmaxf(mx, __shfl_xor(mx, 1, 32));
      mx = fmaxf(mx, __shfl_xor(mx, 2, 32));
      mx = fmaxf(mx, __shfl_xor(mx, 4, 32));
      mx = fmaxf(mx, __shfl_xor(mx, 8, 32));
      const float mnew  = fmaxf(m8[r], mx);
      const float alpha = exp2f(m8[r] - mnew);
      m8[r] = mnew;
      const float p0 = exp2f(s[0][r] - mnew);
      const float p1 = exp2f(s[1][r] - mnew);
      s[0][r] = p0; s[1][r] = p1;
      float ps = p0 + p1;
      ps += __shfl_xor(ps, 1, 32);
      ps += __shfl_xor(ps, 2, 32);
      ps += __shfl_xor(ps, 4, 32);
      ps += __shfl_xor(ps, 8, 32);
      l8[r] = l8[r] * alpha + ps;
#pragma unroll
      for (int t = 0; t < 4; ++t) o[t][r] *= alpha;
    }
    // ---- D-layout -> A-layout via wave-private LDS round trip ----
#pragma unroll
    for (int r = 0; r < 8; ++r) {
      pl[(r + 8 * hf) * 32 + row]      = (_Float16)s[0][r];
      pl[(r + 8 * hf) * 32 + 16 + row] = (_Float16)s[1][r];
    }
    __builtin_amdgcn_wave_barrier();       // LDS is in-order within a wave
    const v16h aP = load_frag_a(pl + row * 32 + hf * 8);
    __builtin_amdgcn_wave_barrier();
    // ---- O += P (16x32) * V (32x64)  (uses *old* V loads) ----
#pragma unroll
    for (int t = 0; t < 4; ++t)
      o[t] = wmma_f16(aP, vcur[t], o[t]);

#pragma unroll
    for (int t = 0; t < 4; ++t) { kcur[t] = kn[t]; vcur[t] = vn[t]; }
  }

  // ---- finalize: divide by l, write ctx [B,S,H*64] as f16 ----
#pragma unroll
  for (int r = 0; r < 8; ++r) {
    const float inv = 1.0f / l8[r];
    const int srow = q0 + r + 8 * hf;
    const int base = (b * S_DIM + srow) * HID + h * DH;
#pragma unroll
    for (int t = 0; t < 4; ++t)
      ctx[base + t * 16 + row] = (_Float16)(o[t][r] * inv);
  }
}

// ---------- kernel 5: output projection ----------
__global__ __launch_bounds__(256) void k_gemm_out(const _Float16* __restrict__ ch,
                                                  const _Float16* __restrict__ woT,
                                                  const float* __restrict__ bo,
                                                  float* __restrict__ out) {
  __shared__ __align__(16) _Float16 bt[2 * BT_BUF];
  const int lane = threadIdx.x & 31, wave = threadIdx.x >> 5;
  const int row = lane & 15, hf = lane >> 4;
  const int m0 = blockIdx.y * 128 + wave * 16;
  const int n0 = blockIdx.x * 64;

  v8f acc[4] = {};
  gemm_core(ch, woT, bt, m0, n0, acc);

#pragma unroll
  for (int t = 0; t < 4; ++t) {
    const int n = n0 + t * 16 + row;
    const float bn = bo[n];
#pragma unroll
    for (int r = 0; r < 8; ++r) {
      const int m = m0 + r + 8 * hf;
      out[m * HID + n] = acc[t][r] + bn;
    }
  }
}

// ---------- host launcher ----------
extern "C" void kernel_launch(void* const* d_in, const int* in_sizes, int n_in,
                              void* d_out, int out_size, void* d_ws, size_t ws_size,
                              hipStream_t stream) {
  const float* x  = (const float*)d_in[0];
  const float* Wq = (const float*)d_in[1];
  const float* bq = (const float*)d_in[2];
  const float* Wk = (const float*)d_in[3];
  const float* bk = (const float*)d_in[4];
  const float* Wv = (const float*)d_in[5];
  const float* bv = (const float*)d_in[6];
  const float* Wo = (const float*)d_in[7];
  const float* bo = (const float*)d_in[8];
  float* out = (float*)d_out;

  char* ws = (char*)d_ws;
  _Float16* xh  = (_Float16*)(ws + ((size_t)0  << 20));  // 8 MB
  _Float16* wqT = (_Float16*)(ws + ((size_t)8  << 20));  // 2 MB
  _Float16* wkT = (_Float16*)(ws + ((size_t)10 << 20));  // 2 MB
  _Float16* wvT = (_Float16*)(ws + ((size_t)12 << 20));  // 2 MB
  _Float16* woT = (_Float16*)(ws + ((size_t)14 << 20));  // 2 MB
  _Float16* Qb  = (_Float16*)(ws + ((size_t)16 << 20));  // 8 MB
  _Float16* Kb  = (_Float16*)(ws + ((size_t)24 << 20));  // 8 MB
  _Float16* Vt  = (_Float16*)(ws + ((size_t)32 << 20));  // 8 MB
  _Float16* ctx = (_Float16*)(ws + ((size_t)40 << 20));  // 8 MB

  const int NX = B_DIM * S_DIM * HID;                    // 4M elements
  k_cvt_x<<<NX / (256 * 4), 256, 0, stream>>>(x, xh, NX);

  dim3 tb(32, 8);
  dim3 tg(HID / 32, HID / 32);
  k_cvt_wT<<<tg, tb, 0, stream>>>(Wq, wqT);
  k_cvt_wT<<<tg, tb, 0, stream>>>(Wk, wkT);
  k_cvt_wT<<<tg, tb, 0, stream>>>(Wv, wvT);
  k_cvt_wT<<<tg, tb, 0, stream>>>(Wo, woT);

  k_gemm_qkv<<<dim3(HID / 64, (B_DIM * S_DIM) / 128), 256, 0, stream>>>(
      xh, wqT, wkT, wvT, bq, bk, bv, Qb, Kb, Vt);

  k_attn<<<dim3(S_DIM / 64, B_DIM * NH), 128, 0, stream>>>(Qb, Kb, Vt, ctx);

  k_gemm_out<<<dim3(HID / 64, (B_DIM * S_DIM) / 128), 256, 0, stream>>>(
      ctx, woT, bo, out);
}
